// R_mix_NAM_DNN_87411174408658
// MI455X (gfx1250) — matrix operand order repaired and checked by reading, not compiled, not run on hardware
//
#include <hip/hip_runtime.h>

typedef __attribute__((ext_vector_type(16))) _Float16 v16h;
typedef __attribute__((ext_vector_type(8)))  _Float16 v8h;
typedef __attribute__((ext_vector_type(8)))  float    v8f;

#define B_ROWS   32768
#define D_IN     128
#define H0       256
#define H1       256
#define NBLK     128
#define HTAU     32

// -------------------------------------------------------------------------
// f32 -> f16 conversion (x and both weight matrices are consumed in f16)
// -------------------------------------------------------------------------
__global__ __launch_bounds__(256)
void cvt_f32_f16_kernel(const float* __restrict__ in, _Float16* __restrict__ out, int n) {
  int i = blockIdx.x * blockDim.x + threadIdx.x;
  if (i < n) out[i] = (_Float16)in[i];
}

// -------------------------------------------------------------------------
// Fused mu branch:
//   h1 = relu(x @ W1^T + b1)       [64-row tile, lives only in LDS]
//   h2 = relu(h1 @ W2^T + b2)      [never materialized]
//   mu = h2 @ w3 + b3              [folded into GEMM2 epilogue via ds_add_f32]
//
// Block: 256 threads = 8 waves, 64-row M tile.
//   wave&3  -> 16-row M subtile, wave>>2 -> N half (128 cols, 8 n-tiles).
// x tile staged via CDNA5 async global->LDS (global_load_async_to_lds_b128,
// s_wait_asynccnt); GEMMs use v_wmma_f32_16x16x32_f16 with f32 accumulate.
// -------------------------------------------------------------------------
__global__ __launch_bounds__(256)
void mu_fused_wmma(const _Float16* __restrict__ Xh,  // [B,128] f16
                   const _Float16* __restrict__ W1h, // [256,128] f16
                   const float*    __restrict__ b1,  // [256]
                   const _Float16* __restrict__ W2h, // [256,256] f16
                   const float*    __restrict__ b2,  // [256]
                   const float*    __restrict__ w3,  // [256]
                   const float*    __restrict__ b3,  // [1]
                   float*          __restrict__ mu)  // [B]
{
  constexpr int LDX = D_IN + 8;   // 136 halfs -> 272B row stride (16B aligned, no bank conflicts)
  constexpr int LDH = H0 + 8;     // 264 halfs -> 528B row stride
  __shared__ __align__(16) _Float16 Xs[64 * LDX];
  __shared__ __align__(16) _Float16 Hs[64 * LDH];
  __shared__ float muAcc[64];

  const int tid  = threadIdx.x;
  const int wave = tid >> 5;
  const int lane = tid & 31;
  const int m_block = blockIdx.x * 64;

  if (tid < 64) muAcc[tid] = 0.0f;

  // ---- async-stage x tile (64 x 128 halfs) into LDS: 1024 16B chunks ----
  {
    constexpr int chunksPerRow = D_IN / 8;            // 16
    for (int c = tid; c < 64 * chunksPerRow; c += 256) {
      int r  = c / chunksPerRow;
      int kc = (c % chunksPerRow) * 8;
      unsigned lds_addr = (unsigned)(size_t)(&Xs[r * LDX + kc]);
      const _Float16* gp = &Xh[(size_t)(m_block + r) * D_IN + kc];
      asm volatile("global_load_async_to_lds_b128 %0, %1, off"
                   :: "v"(lds_addr), "v"(gp) : "memory");
    }
    asm volatile("s_wait_asynccnt 0x0" ::: "memory");
  }
  __syncthreads();

  const int m_sub   = (wave & 3) * 16;
  const int n_half  = wave >> 2;          // 0 or 1
  const int half_id = lane >> 4;
  const int lane16  = lane & 15;
  const int row0    = m_sub + 8 * half_id;   // D-layout base row within tile

  // ================= GEMM1: h1 tile (LDS) = relu(x @ W1^T + b1) =================
  // ISA A layout (16-bit 16x32): elems 0-7 -> K=8*half+0..7, elems 8-15 -> K=16+8*half+0..7
  v16h a1[D_IN / 32];
  {
    const _Float16* ap = &Xs[(m_sub + lane16) * LDX];
    #pragma unroll
    for (int ks = 0; ks < D_IN / 32; ++ks) {
      v8h lo = *(const v8h*)(ap + ks * 32 + 8 * half_id);
      v8h hi = *(const v8h*)(ap + ks * 32 + 16 + 8 * half_id);
      a1[ks] = __builtin_shufflevector(lo, hi, 0,1,2,3,4,5,6,7,8,9,10,11,12,13,14,15);
    }
  }

  for (int t = 0; t < 8; ++t) {
    const int n = n_half * 128 + t * 16 + lane16;   // output column
    v8f acc = {};
    #pragma unroll
    for (int ks = 0; ks < D_IN / 32; ++ks) {
      // B layout (32x16): lane -> col n, elems 0-7 -> K = 8*half + 0..7, elems 8-15 -> K = 16 + 8*half + 0..7
      const _Float16* wr = &W1h[(size_t)n * D_IN + ks * 32 + 8 * half_id];
      v8h blo = *(const v8h*)(wr);
      v8h bhi = *(const v8h*)(wr + 16);
      v16h bf = __builtin_shufflevector(blo, bhi, 0,1,2,3,4,5,6,7,8,9,10,11,12,13,14,15);
      acc = __builtin_amdgcn_wmma_f32_16x16x32_f16(false, a1[ks], false, bf,
                                                   (short)0, acc, false, false);
    }
    const float bv = b1[n];
    #pragma unroll
    for (int r = 0; r < 8; ++r) {      // D: VGPR r -> row row0+r, col n
      float v = acc[r] + bv;
      v = v > 0.0f ? v : 0.0f;
      Hs[(row0 + r) * LDH + n] = (_Float16)v;
    }
  }
  __syncthreads();

  // ============ GEMM2 + mu: h2 = relu(h1 @ W2^T + b2); mu += h2 * w3 ============
  v16h a2[H0 / 32];
  {
    const _Float16* hp = &Hs[(m_sub + lane16) * LDH];
    #pragma unroll
    for (int ks = 0; ks < H0 / 32; ++ks) {
      v8h lo = *(const v8h*)(hp + ks * 32 + 8 * half_id);
      v8h hi = *(const v8h*)(hp + ks * 32 + 16 + 8 * half_id);
      a2[ks] = __builtin_shufflevector(lo, hi, 0,1,2,3,4,5,6,7,8,9,10,11,12,13,14,15);
    }
  }

  float mupart[8];
  #pragma unroll
  for (int r = 0; r < 8; ++r) mupart[r] = 0.0f;

  for (int t = 0; t < 8; ++t) {
    const int n = n_half * 128 + t * 16 + lane16;
    v8f acc = {};
    #pragma unroll
    for (int ks = 0; ks < H0 / 32; ++ks) {
      const _Float16* wr = &W2h[(size_t)n * H0 + ks * 32 + 8 * half_id];
      v8h blo = *(const v8h*)(wr);
      v8h bhi = *(const v8h*)(wr + 16);
      v16h bf = __builtin_shufflevector(blo, bhi, 0,1,2,3,4,5,6,7,8,9,10,11,12,13,14,15);
      acc = __builtin_amdgcn_wmma_f32_16x16x32_f16(false, a2[ks], false, bf,
                                                   (short)0, acc, false, false);
    }
    const float bv  = b2[n];
    const float w3n = w3[n];
    #pragma unroll
    for (int r = 0; r < 8; ++r) {
      float v = acc[r] + bv;
      v = v > 0.0f ? v : 0.0f;
      mupart[r] = fmaf(v, w3n, mupart[r]);   // rows are t-invariant: accumulate in regs
    }
  }
  #pragma unroll
  for (int r = 0; r < 8; ++r)
    atomicAdd(&muAcc[row0 + r], mupart[r]);   // ds_add_f32
  __syncthreads();

  if (tid < 64) mu[m_block + tid] = muAcc[tid] + b3[0];
}

// -------------------------------------------------------------------------
// tau branch: one block (128 threads) per batch row; thread n handles
// block-feature n: 32 softplus + weighted reduce, then LDS reduce over n.
// softplus(z) = max(z,0) + log(1 + exp(-|z|))   (stable)
// -------------------------------------------------------------------------
__global__ __launch_bounds__(128)
void tau_kernel(const float* __restrict__ x,      // [B,128]
                const float* __restrict__ w_t1,   // [128,32]
                const float* __restrict__ b_t1,   // [128,32]
                const float* __restrict__ w_t2,   // [128,1,32]
                const float* __restrict__ b_t2,   // [128,1]
                const float* __restrict__ w_lr,   // [1,128]
                const float* __restrict__ b_lr,   // [1]
                float*       __restrict__ out_lr) // [B]
{
  __shared__ float red[NBLK];
  const int b = blockIdx.x;
  const int n = threadIdx.x;

  const float xv = x[(size_t)b * NBLK + n];
  const float* w1  = &w_t1[n * HTAU];
  const float* bb1 = &b_t1[n * HTAU];
  const float* w2  = &w_t2[n * HTAU];

  float acc = 0.0f;
  #pragma unroll
  for (int h = 0; h < HTAU; ++h) {
    float z  = fmaf(xv, w1[h], bb1[h]);
    float sp = fmaxf(z, 0.0f) + __logf(1.0f + __expf(-fabsf(z)));
    acc = fmaf(sp, w2[h], acc);
  }
  red[n] = (acc + b_t2[n]) * w_lr[n];
  __syncthreads();

  #pragma unroll
  for (int s = 64; s >= 1; s >>= 1) {
    if (n < s) red[n] += red[n + s];
    __syncthreads();
  }
  if (n == 0) out_lr[b] = red[0] + b_lr[0];
}

// -------------------------------------------------------------------------
// launch
// -------------------------------------------------------------------------
extern "C" void kernel_launch(void* const* d_in, const int* in_sizes, int n_in,
                              void* d_out, int out_size, void* d_ws, size_t ws_size,
                              hipStream_t stream) {
  const float* x     = (const float*)d_in[0];
  const float* w_mu1 = (const float*)d_in[1];
  const float* b_mu1 = (const float*)d_in[2];
  const float* w_mu2 = (const float*)d_in[3];
  const float* b_mu2 = (const float*)d_in[4];
  const float* w_mu3 = (const float*)d_in[5];
  const float* b_mu3 = (const float*)d_in[6];
  const float* w_t1  = (const float*)d_in[7];
  const float* b_t1  = (const float*)d_in[8];
  const float* w_t2  = (const float*)d_in[9];
  const float* b_t2  = (const float*)d_in[10];
  const float* w_lr  = (const float*)d_in[11];
  const float* b_lr  = (const float*)d_in[12];

  float* mu     = (float*)d_out;            // [B]
  float* out_lr = (float*)d_out + B_ROWS;   // [B]

  // workspace (f16): x_h (8MB) | W1h (64KB) | W2h (128KB)
  _Float16* xh  = (_Float16*)d_ws;
  _Float16* w1h = xh  + (size_t)B_ROWS * D_IN;
  _Float16* w2h = w1h + (size_t)H0 * D_IN;

  // 1) f32 -> f16 conversions (x, W1, W2)
  cvt_f32_f16_kernel<<<(B_ROWS * D_IN) / 256, 256, 0, stream>>>(x, xh, B_ROWS * D_IN);
  cvt_f32_f16_kernel<<<(H0 * D_IN) / 256, 256, 0, stream>>>(w_mu1, w1h, H0 * D_IN);
  cvt_f32_f16_kernel<<<(H1 * H0) / 256, 256, 0, stream>>>(w_mu2, w2h, H1 * H0);

  // 2) fused mu branch (both GEMMs + final dot), one kernel
  mu_fused_wmma<<<B_ROWS / 64, 256, 0, stream>>>(xh, w1h, b_mu1, w2h, b_mu2,
                                                 w_mu3, b_mu3, mu);

  // 3) tau branch -> out_lr
  tau_kernel<<<B_ROWS, 128, 0, stream>>>(x, w_t1, b_t1, w_t2, b_t2, w_lr, b_lr, out_lr);
}
